// GATAgg_28475633173145
// MI455X (gfx1250) — compile-verified, matrix operand order checked by reading, not run on hardware
//
#include <hip/hip_runtime.h>
#include <hip/hip_bf16.h>

// ---------------------------------------------------------------------------
// GAT aggregation, restructured (roofline: ~0.87 GB traffic -> ~38 us @ 23.3 TB/s):
//   v1 = W @ atn_w1, v2 = W @ atn_w2   (attention never needs H = A@W)
//   per node: f[k] = a_k.v2 (+ a_0.v1), tanh, softmax, m = sum coef[k]*a_k
//   out = relu(M @ W + bias) via V_WMMA_F32_16X16X32_BF16 (fp32 accumulate)
// Data movement: neigh_vecs streamed HBM->LDS with the Tensor Data Mover
// (TENSOR_LOAD_TO_LDS, double buffered, s_wait_tensorcnt pipeline).
// ---------------------------------------------------------------------------

typedef __attribute__((ext_vector_type(16))) __bf16       v16bf;
typedef __attribute__((ext_vector_type(8)))  float        v8f;
typedef __attribute__((ext_vector_type(4)))  unsigned int u32x4;
typedef __attribute__((ext_vector_type(8)))  int          i32x8;
typedef __attribute__((ext_vector_type(4)))  int          i32x4;

#define WAVES 2
#define TPB   (WAVES * 32)
#define NODE_BUF_FLOATS (32 * 128)          // one node's neighbor block, 16 KB

__device__ __forceinline__ unsigned short f2bf(float x) {
    unsigned int u = __float_as_uint(x);
    u += 0x7FFFu + ((u >> 16) & 1u);        // round-to-nearest-even
    return (unsigned short)(u >> 16);
}

__device__ __forceinline__ float wsum(float v) {
#pragma unroll
    for (int off = 16; off; off >>= 1) v += __shfl_xor(v, off, 32);
    return v;                                // all 32 lanes get the total
}

__device__ __forceinline__ float dot4(float4 a, float4 b) {
    return a.x * b.x + a.y * b.y + a.z * b.z + a.w * b.w;
}

union FragU { uint4 u[2]; v16bf v; };

// TDM 2D tile load: 32 rows x 128 fp32 (tensor_dim0=128, stride 128) -> LDS.
// D# layout per CDNA5 ISA 8.3/8.4: group0 = {count/type/lds_addr/global_addr},
// group1 = {data_size, dims, tile dims, strides}; groups 2/3 zero (2D tensor).
// 6-arg toolchain builtin: (u32x4 g0, i32x8 g1, i32x4 g2, i32x4 g3, i32x8, i32 cpol)
__device__ __forceinline__ void tdm_load_rows(const float* gsrc, unsigned lds_off) {
    unsigned long long ga = (unsigned long long)(const void*)gsrc;
    u32x4 g0;
    g0[0] = 1u;                                        // count=1, user descriptor
    g0[1] = lds_off;                                   // lds_addr (bytes)
    g0[2] = (unsigned int)ga;                          // global_addr[31:0]
    g0[3] = (unsigned int)(ga >> 32) | 0x80000000u;    // global_addr[56:32] | type=2
    i32x8 g1;
    g1[0] = 0x00020000;          // workgroup_mask=0, data_size=2 (4 bytes)
    g1[1] = (int)(128u << 16);   // tensor_dim0 = 128 (elements)
    g1[2] = (int)(32u  << 16);   // tensor_dim1 = 32
    g1[3] = (int)(128u << 16);   // tile_dim0 = 128
    g1[4] = 32;                  // tile_dim1 = 32, tile_dim2 = 0
    g1[5] = 128;                 // tensor_dim0_stride = 128
    g1[6] = 0;                   // tensor_dim1_stride unused (2D)
    g1[7] = 0;
    i32x4 gz4 = {0, 0, 0, 0};
    i32x8 gz8 = {0, 0, 0, 0, 0, 0, 0, 0};
    __builtin_amdgcn_tensor_load_to_lds(g0, g1, gz4, gz4, gz8, 0);
}

__global__ __launch_bounds__(TPB) void gat_agg_kernel(
    const float* __restrict__ selfv,   // [N,128]
    const float* __restrict__ neigh,   // [N,32,128]
    const float* __restrict__ W,       // [128,128] row-major (k, n)
    const float* __restrict__ w1,      // [128]
    const float* __restrict__ w2,      // [128]
    const float* __restrict__ b1p,
    const float* __restrict__ b2p,
    const float* __restrict__ biasv,   // [128]
    float*       __restrict__ out,     // [N,128]
    int N)
{
    // W in WMMA B-fragment layout: [kt(4)][nt(8)][lane(32)][16 bf16] = 32 KB
    __shared__ __attribute__((aligned(32))) unsigned short wfrag[4 * 8 * 32 * 16];
    // per-wave m staging tile, row-major bf16 [16 nodes][128]
    __shared__ __attribute__((aligned(32))) unsigned short mst[WAVES][16][128];
    __shared__ float v1s[128], v2s[128];
    // dynamic LDS: per-wave double-buffered TDM node buffers [WAVES][2][4096] f32
    extern __shared__ __attribute__((aligned(16))) float nbuf_dyn[];

    const int tid  = threadIdx.x;
    const int w    = tid >> 5;
    const int lane = tid & 31;

    // ---- stage W into B-fragment layout (bf16), ISA 16-bit B layout:
    // lane L holds column n = nt*16 + (L&15); K slots {kb..kb+7, kb+16..kb+23},
    // kb = kt*32 + (L>=16 ? 8 : 0)
    for (int e = tid; e < 4 * 8 * 32; e += TPB) {
        const int kt = e >> 8;
        const int nt = (e >> 5) & 7;
        const int ln = e & 31;
        const int n  = nt * 16 + (ln & 15);
        const int kb = kt * 32 + ((ln & 16) ? 8 : 0);
        unsigned short* dst = &wfrag[e * 16];
#pragma unroll
        for (int s = 0; s < 8; ++s) dst[s]     = f2bf(W[(kb + s) * 128 + n]);
#pragma unroll
        for (int s = 0; s < 8; ++s) dst[8 + s] = f2bf(W[(kb + 16 + s) * 128 + n]);
    }
    // ---- v1 = W@w1, v2 = W@w2 (hits L2, negligible)
    for (int k = tid; k < 128; k += TPB) {
        float s1 = 0.f, s2 = 0.f;
        for (int n = 0; n < 128; ++n) {
            const float wv = W[k * 128 + n];
            s1 += wv * w1[n];
            s2 += wv * w2[n];
        }
        v1s[k] = s1; v2s[k] = s2;
    }
    __syncthreads();

    const float4 v1seg = *(const float4*)&v1s[4 * lane];
    const float4 v2seg = *(const float4*)&v2s[4 * lane];
    const float  b12   = b1p[0] + b2p[0];

    float* mybuf = nbuf_dyn + w * 2 * NODE_BUF_FLOATS;
    const unsigned lds_off0 = (unsigned)(unsigned long long)(const void*)mybuf;

    const int numTiles = (N + 15) >> 4;
    for (int tile = blockIdx.x * WAVES + w; tile < numTiles; tile += gridDim.x * WAVES) {
        const int tb = tile << 4;

        // prime the TDM pipeline: node tb -> buffer 0
        tdm_load_rows(neigh + (size_t)tb * 4096, lds_off0);

        // ================= Phase A: attention + weighted row sum ============
        for (int j = 0; j < 16; ++j) {            // wave-uniform loop
            const int node = tb + j;
            // issue next node's DMA into the other buffer, then wait for ours
            if (j < 15 && node + 1 < N) {
                tdm_load_rows(neigh + (size_t)(node + 1) * 4096,
                              lds_off0 + (unsigned)(((j + 1) & 1) * NODE_BUF_FLOATS * 4));
                __builtin_amdgcn_s_wait_tensorcnt(1);
            } else {
                __builtin_amdgcn_s_wait_tensorcnt(0);
            }

            unsigned int* dst = (unsigned int*)&mst[w][j][4 * lane];
            if (node < N) {                       // wave-uniform branch
                const float* buf = mybuf + (j & 1) * NODE_BUF_FLOATS;

                float4 r[33];
                float  f[33];
                r[0] = ((const float4*)(selfv + (size_t)node * 128))[lane];
#pragma unroll
                for (int k = 1; k < 33; ++k)
                    r[k] = *(const float4*)&buf[(k - 1) * 128 + 4 * lane];

                const float f1 = wsum(dot4(r[0], v1seg));
#pragma unroll
                for (int k = 0; k < 33; ++k)
                    f[k] = wsum(dot4(r[k], v2seg));

                float mx = -1e30f;
#pragma unroll
                for (int k = 0; k < 33; ++k) {
                    f[k] = tanhf(f1 + f[k] + b12);   // TEMP == 1.0
                    mx = fmaxf(mx, f[k]);
                }
                float se = 0.f;
#pragma unroll
                for (int k = 0; k < 33; ++k) { f[k] = __expf(f[k] - mx); se += f[k]; }
                const float inv = 1.f / se;

                float4 m = make_float4(0.f, 0.f, 0.f, 0.f);
#pragma unroll
                for (int k = 0; k < 33; ++k) {
                    const float c = f[k] * inv;
                    m.x += c * r[k].x; m.y += c * r[k].y;
                    m.z += c * r[k].z; m.w += c * r[k].w;
                }
                dst[0] = (unsigned int)f2bf(m.x) | ((unsigned int)f2bf(m.y) << 16);
                dst[1] = (unsigned int)f2bf(m.z) | ((unsigned int)f2bf(m.w) << 16);
            } else {
                dst[0] = 0u; dst[1] = 0u;
            }
        }

        // ================= Phase B: 16x128 @ 128x128 via WMMA ==============
        // A-fragment (16-bit, 16x32): lane L -> row L&15,
        // K slots {c0..c0+7, c0+16..c0+23}, c0 = kt*32 + (L>=16 ? 8 : 0)
        FragU A[4];
        const int row = lane & 15;
#pragma unroll
        for (int kt = 0; kt < 4; ++kt) {
            const int c0 = kt * 32 + ((lane & 16) ? 8 : 0);
            A[kt].u[0] = *(const uint4*)&mst[w][row][c0];
            A[kt].u[1] = *(const uint4*)&mst[w][row][c0 + 16];
        }
        const bool fullTile = (tb + 16 <= N);     // uniform fast path
#pragma unroll
        for (int nt = 0; nt < 8; ++nt) {
            v8f acc = {0.f, 0.f, 0.f, 0.f, 0.f, 0.f, 0.f, 0.f};
#pragma unroll
            for (int kt = 0; kt < 4; ++kt) {
                const v16bf b = *(const v16bf*)&wfrag[((kt << 3) + nt) * 512 + lane * 16];
                acc = __builtin_amdgcn_wmma_f32_16x16x32_bf16(
                          false, A[kt].v, false, b, (short)0, acc, false, false);
            }
            // C/D layout: lane L, vgpr i -> row i + (L>=16 ? 8:0), col nt*16+(L&15)
            const int   col  = nt * 16 + (lane & 15);
            const float bv   = biasv[col];
            const int   rb   = tb + ((lane & 16) ? 8 : 0);
            if (fullTile) {
#pragma unroll
                for (int i = 0; i < 8; ++i)
                    out[(size_t)(rb + i) * 128 + col] = fmaxf(acc[i] + bv, 0.f);
            } else {
#pragma unroll
                for (int i = 0; i < 8; ++i)
                    if (rb + i < N)
                        out[(size_t)(rb + i) * 128 + col] = fmaxf(acc[i] + bv, 0.f);
            }
        }
    }
}

extern "C" void kernel_launch(void* const* d_in, const int* in_sizes, int n_in,
                              void* d_out, int out_size, void* d_ws, size_t ws_size,
                              hipStream_t stream) {
    const float* selfv = (const float*)d_in[0];   // [N,128]
    const float* neigh = (const float*)d_in[1];   // [N,32,128]
    // d_in[2] = n_sample (fixed S=32 per reference shapes)
    const float* W     = (const float*)d_in[3];   // [128,128]
    const float* w1    = (const float*)d_in[4];   // [128,1]
    const float* w2    = (const float*)d_in[5];   // [128,1]
    const float* b1    = (const float*)d_in[6];
    const float* b2    = (const float*)d_in[7];
    const float* bv    = (const float*)d_in[8];   // [128]
    float* out = (float*)d_out;

    const int N        = in_sizes[0] / 128;
    const int numTiles = (N + 15) / 16;
    const int blocks   = (numTiles + WAVES - 1) / WAVES;
    const size_t dynLds = (size_t)WAVES * 2 * NODE_BUF_FLOATS * sizeof(float); // 64 KB

    gat_agg_kernel<<<blocks, TPB, dynLds, stream>>>(selfv, neigh, W, w1, w2, b1, b2, bv, out, N);
}
